// GPT2_88098369176362
// MI455X (gfx1250) — matrix operand order repaired and checked by reading, not compiled, not run on hardware
//
#include <hip/hip_runtime.h>
#include <hip/hip_bf16.h>
#include <math.h>

// GPT-2 forward on gfx1250 (MI455X).
// f16 weights stored TRANSPOSED [N,K]: every GEMM computes D^T = Wt x Act^T
// with both WMMA fragments as contiguous global b128 (no LDS, no barriers).
// Attention: flash WMMA with transposed-V cache; softmax row-sums via a WMMA
// against a ones matrix; only the max reduction uses cross-lane shuffles.

typedef __attribute__((ext_vector_type(16))) _Float16 v16h;
typedef __attribute__((ext_vector_type(8)))  _Float16 v8h;
typedef __attribute__((ext_vector_type(8)))  float    v8f;

union Frag16 { v16h v; v8h h[2]; };

#define WMMA_F16(a, b, c) \
  __builtin_amdgcn_wmma_f32_16x16x32_f16(false, (a), false, (b), (short)0, (c), false, false)

constexpr int Bc = 2, Tt = 1024, Cc = 768, Hh = 12, Dd = 64, Ll = 12, FFf = 3072, Vv = 50257;

// ---------------- fp32 [L,K,N] -> f16 transposed [L,N,K]; grid (K/256, N, L)
__global__ void k_cvtT(const float* __restrict__ s, _Float16* __restrict__ d, int K, int N) {
  const int  k = blockIdx.x * 256 + threadIdx.x;
  const int  n = blockIdx.y;
  const long l = blockIdx.z;
  if (k >= K) return;
  d[(l * N + n) * (long)K + k] = (_Float16)s[l * (long)K * N + (long)k * N + n];
}

// ---------------- V transpose: qkv[b,t, 2C + h*D + d] -> vT[(b*H+h)*D + d][t]
__global__ void k_vT(const _Float16* __restrict__ qkv, _Float16* __restrict__ vT) {
  const int t  = blockIdx.x * 256 + threadIdx.x;  // grid.x = T/256
  const int d  = blockIdx.y;                      // D
  const int bh = blockIdx.z;                      // B*H
  const int b = bh / Hh, hh = bh % Hh;
  vT[((long)bh * Dd + d) * Tt + t] =
      qkv[(long)(b * Tt + t) * (3 * Cc) + 2 * Cc + hh * Dd + d];
}

// ---------------------------------------------------------------- embedding
__global__ void k_embed(const int* __restrict__ idx, const float* __restrict__ tok,
                        const float* __restrict__ pos, float* __restrict__ x) {
  int tid = blockIdx.x * blockDim.x + threadIdx.x;
  const int total = Bc * Tt * Cc;
  if (tid >= total) return;
  int c  = tid % Cc;
  int bt = tid / Cc;
  int t  = bt % Tt;
  x[tid] = tok[(long)idx[bt] * Cc + c] + pos[t * Cc + c];
}

// ---------------- layernorm: one wave per row, shuffle reduce, f16 out
__global__ __launch_bounds__(256) void k_ln(const float* __restrict__ x,
                                            const float* __restrict__ s,
                                            const float* __restrict__ b,
                                            _Float16* __restrict__ o) {
  const int lane = threadIdx.x & 31;
  const int row  = blockIdx.x * 8 + (threadIdx.x >> 5);
  const float* xr = x + (long)row * Cc;
  float v[Cc / 32];
  float sum = 0.f;
#pragma unroll
  for (int i = 0; i < Cc / 32; ++i) { v[i] = xr[lane + i * 32]; sum += v[i]; }
#pragma unroll
  for (int m = 1; m < 32; m <<= 1) sum += __shfl_xor(sum, m, 32);
  const float mu = sum * (1.0f / Cc);
  float var = 0.f;
#pragma unroll
  for (int i = 0; i < Cc / 32; ++i) { float d = v[i] - mu; var += d * d; }
#pragma unroll
  for (int m = 1; m < 32; m <<= 1) var += __shfl_xor(var, m, 32);
  const float rstd = rsqrtf(var * (1.0f / Cc) + 1e-5f);
  _Float16* orow = o + (long)row * Cc;
#pragma unroll
  for (int i = 0; i < Cc / 32; ++i) {
    int c = lane + i * 32;
    orow[c] = (_Float16)((v[i] - mu) * rstd * s[c] + b[c]);
  }
}

// ---------------------------------------------------------------- WMMA GEMM, LDS-free
// Out[m,n] computed as D^T = Wt[N,K] (A-frags) x Act[M,K] (B-frags = Act^T).
// 8 independent waves/block; wave tile 64n x 64m = 16 wmma per 32-k step.
// FUSE: 0 = bias, 1 = bias + fp32 residual, 2 = bias + exact GELU
// FAST: true  => N is a multiple of 128 (no guards, vector epilogue only)
//       false => guarded scalar epilogue only (LM head)
template <int FUSE, bool FAST, typename OutT>
__global__ __launch_bounds__(256) void k_gemm(const _Float16* __restrict__ Wt,
                                              const _Float16* __restrict__ Act,
                                              const float* __restrict__ bias,
                                              const float* __restrict__ resid,
                                              OutT* __restrict__ Co,
                                              int M, int N, int K) {
  const int lane = threadIdx.x & 31, wave = threadIdx.x >> 5;
  const int half = lane >> 4, l16 = lane & 15;
  const int n0 = blockIdx.x * 128 + (wave & 1) * 64;   // 4 n-subtiles of 16
  const int m0 = blockIdx.y * 256 + (wave >> 1) * 64;  // 4 m-subtiles of 16

  v8f c[4][4] = {};
  const _Float16* ap[4];
  const _Float16* bp[4];
#pragma unroll
  for (int i = 0; i < 4; ++i) {
    int nr = n0 + i * 16 + l16;
    if (!FAST) nr = nr < N ? nr : N - 1;
    ap[i] = Wt + (long)nr * K + half * 8;                      // + k0 / + k0+16
  }
#pragma unroll
  for (int j = 0; j < 4; ++j)
    bp[j] = Act + (long)(m0 + j * 16 + l16) * K + half * 8;    // + k0 / + k0+16

  for (int k0 = 0; k0 < K; k0 += 32) {
    Frag16 a[4];
#pragma unroll
    for (int i = 0; i < 4; ++i) {
      a[i].h[0] = *(const v8h*)(ap[i] + k0);
      a[i].h[1] = *(const v8h*)(ap[i] + k0 + 16);
    }
#pragma unroll
    for (int j = 0; j < 4; ++j) {
      Frag16 b;
      b.h[0] = *(const v8h*)(bp[j] + k0);
      b.h[1] = *(const v8h*)(bp[j] + k0 + 16);
#pragma unroll
      for (int i = 0; i < 4; ++i)
        c[i][j] = WMMA_F16(a[i].v, b.v, c[i][j]);
    }
  }

  // epilogue: lane holds col m; elem r -> n = nbase + r (contiguous)
  if constexpr (FAST) {
#pragma unroll
    for (int i = 0; i < 4; ++i) {
      const int nb = n0 + i * 16 + 8 * half;
      float bv[8];
      if (bias) {
        float4 q0 = *(const float4*)(bias + nb);
        float4 q1 = *(const float4*)(bias + nb + 4);
        bv[0]=q0.x; bv[1]=q0.y; bv[2]=q0.z; bv[3]=q0.w;
        bv[4]=q1.x; bv[5]=q1.y; bv[6]=q1.z; bv[7]=q1.w;
      } else {
#pragma unroll
        for (int r = 0; r < 8; ++r) bv[r] = 0.f;
      }
#pragma unroll
      for (int j = 0; j < 4; ++j) {
        const long mrow = m0 + j * 16 + l16;
        float o[8];
#pragma unroll
        for (int r = 0; r < 8; ++r) o[r] = c[i][j][r] + bv[r];
        if constexpr (FUSE == 1) {
          float4 r0 = *(const float4*)(resid + mrow * N + nb);
          float4 r1 = *(const float4*)(resid + mrow * N + nb + 4);
          o[0]+=r0.x; o[1]+=r0.y; o[2]+=r0.z; o[3]+=r0.w;
          o[4]+=r1.x; o[5]+=r1.y; o[6]+=r1.z; o[7]+=r1.w;
        }
        if constexpr (FUSE == 2) {
#pragma unroll
          for (int r = 0; r < 8; ++r)
            o[r] = 0.5f * o[r] * (1.f + erff(o[r] * 0.70710678118654752f));
        }
        if constexpr (sizeof(OutT) == 2) {
          v8h pk;
#pragma unroll
          for (int r = 0; r < 8; ++r) pk[r] = (_Float16)o[r];
          *(v8h*)((_Float16*)Co + mrow * N + nb) = pk;
        } else {
          float4 s0 = {o[0], o[1], o[2], o[3]};
          float4 s1 = {o[4], o[5], o[6], o[7]};
          *(float4*)((float*)Co + mrow * N + nb)     = s0;
          *(float4*)((float*)Co + mrow * N + nb + 4) = s1;
        }
      }
    }
  } else {
#pragma unroll
    for (int i = 0; i < 4; ++i) {
      const int nb = n0 + i * 16 + 8 * half;
#pragma unroll
      for (int j = 0; j < 4; ++j) {
        const long mrow = m0 + j * 16 + l16;
#pragma unroll
        for (int r = 0; r < 8; ++r) {
          const int n = nb + r;
          if (n < N) {
            float v = c[i][j][r] + (bias ? bias[n] : 0.f);
            if constexpr (FUSE == 1) v += resid[mrow * N + n];
            if constexpr (FUSE == 2) v = 0.5f * v * (1.f + erff(v * 0.70710678118654752f));
            Co[mrow * N + n] = (OutT)v;
          }
        }
      }
    }
  }
}

// ---------------------------------------------------------------- flash attention
// one wave per 16-query tile, 8 independent waves/block; 32 keys per chunk;
// only the diagonal chunk masked; V from transposed cache (b128 frags);
// softmax row-sum via WMMA against a ones matrix (no sum shuffles).
__global__ __launch_bounds__(256) void k_attn(const _Float16* __restrict__ qkv,
                                              const _Float16* __restrict__ vT,
                                              _Float16* __restrict__ y) {
  const int wave = threadIdx.x >> 5, lane = threadIdx.x & 31;
  const int gt = blockIdx.x * 8 + wave;          // 0 .. B*H*64-1
  const int qt = gt & 63;
  const int hh = (gt >> 6) % Hh;
  const int bb = gt / (64 * Hh);
  const int half = lane >> 4, l16 = lane & 15;
  const float scale = 0.125f;
  __shared__ __align__(16) _Float16 Ps[8][16][32];
  _Float16 (*P)[32] = Ps[wave];

  Frag16 qa[2], ones;
#pragma unroll
  for (int e = 0; e < 16; ++e) ones.v[e] = (_Float16)1.f;
  {
    const v8h* qp = (const v8h*)(qkv + (long)(bb * Tt + qt * 16 + l16) * (3 * Cc) + hh * Dd);
#pragma unroll
    for (int ch = 0; ch < 2; ++ch) {
      qa[ch].h[0] = qp[ch * 4 + half];
      qa[ch].h[1] = qp[ch * 4 + 2 + half];
    }
  }
  const _Float16* vbase = vT + ((long)(bb * Hh + hh) * Dd) * Tt;

  float mrow[8], lrow[8];
  v8f acc[4] = {{}, {}, {}, {}};
#pragma unroll
  for (int r = 0; r < 8; ++r) { mrow[r] = -1e30f; lrow[r] = 0.f; }

  const int nfull = qt >> 1;                     // chunk nfull is the diagonal
  for (int ck = 0; ck <= nfull; ++ck) {
    const bool masked = (ck == nfull);           // wave-uniform

    // ---- S tiles (two 16-key subtiles)
    v8f s[2] = {{}, {}};
#pragma unroll
    for (int t = 0; t < 2; ++t) {
      const int key = ck * 32 + t * 16 + l16;
      const v8h* kp = (const v8h*)(qkv + (long)(bb * Tt + key) * (3 * Cc) + Cc + hh * Dd);
#pragma unroll
      for (int ch = 0; ch < 2; ++ch) {
        Frag16 kf;
        kf.h[0] = kp[ch * 4 + half];
        kf.h[1] = kp[ch * 4 + 2 + half];
        s[t] = WMMA_F16(qa[ch].v, kf.v, s[t]);
      }
    }

    // ---- running max + exp (only max needs cross-lane shuffles)
    float p0[8], p1[8], alpha[8];
#pragma unroll
    for (int r = 0; r < 8; ++r) {
      float a0 = s[0][r] * scale;
      float a1 = s[1][r] * scale;
      if (masked) {
        const int qg = qt * 16 + r + half * 8;
        const int k0g = ck * 32 + l16;
        a0 = (k0g > qg)      ? -1e30f : a0;
        a1 = (k0g + 16 > qg) ? -1e30f : a1;
      }
      float v = fmaxf(a0, a1);
      for (int m = 1; m <= 8; m <<= 1) v = fmaxf(v, __shfl_xor(v, m, 32));
      const float mnew = fmaxf(mrow[r], v);
      alpha[r] = __expf(mrow[r] - mnew);
      mrow[r]  = mnew;
      p0[r] = __expf(a0 - mnew);
      p1[r] = __expf(a1 - mnew);
    }

    // ---- stage P (16x32) to LDS, reload in A-fragment layout
#pragma unroll
    for (int r = 0; r < 8; ++r) {
      P[r + half * 8][l16]      = (_Float16)p0[r];
      P[r + half * 8][16 + l16] = (_Float16)p1[r];
    }
    Frag16 pa;
    {
      const v8h* pp = (const v8h*)&P[l16][0];
      pa.h[0] = pp[half];
      pa.h[1] = pp[2 + half];
    }

    // ---- row sums via WMMA: P @ ones(32x16) replicates rowsum over cols
    {
      v8f zz = {};
      v8f ssum = WMMA_F16(pa.v, ones.v, zz);
#pragma unroll
      for (int r = 0; r < 8; ++r) lrow[r] = lrow[r] * alpha[r] + ssum[r];
    }

    // ---- acc = acc*alpha + P @ V ; V frag: keys 8*half+0..7 and 16+8*half+0..7 per lane
#pragma unroll
    for (int dt = 0; dt < 4; ++dt) {
#pragma unroll
      for (int r = 0; r < 8; ++r) acc[dt][r] *= alpha[r];
      Frag16 vf;
      const v8h* vp = (const v8h*)(vbase + (long)(dt * 16 + l16) * Tt + ck * 32);
      vf.h[0] = vp[half];
      vf.h[1] = vp[2 + half];
      acc[dt] = WMMA_F16(pa.v, vf.v, acc[dt]);
    }
  }

  // ---- y[b, q, h*64 + d] = acc / l
#pragma unroll
  for (int dt = 0; dt < 4; ++dt)
#pragma unroll
    for (int r = 0; r < 8; ++r) {
      const int q = qt * 16 + r + half * 8;
      const int d = dt * 16 + l16;
      y[(long)(bb * Tt + q) * Cc + hh * Dd + d] = (_Float16)(acc[dt][r] / lrow[r]);
    }
}

// ---------------------------------------------------------------- driver
extern "C" void kernel_launch(void* const* d_in, const int* in_sizes, int n_in,
                              void* d_out, int out_size, void* d_ws, size_t ws_size,
                              hipStream_t stream) {
  const int*   idx   = (const int*)d_in[0];
  const float* tok   = (const float*)d_in[1];
  const float* pos   = (const float*)d_in[2];
  const float* ln1s  = (const float*)d_in[3];
  const float* ln1b  = (const float*)d_in[4];
  const float* Wqkv  = (const float*)d_in[5];
  const float* bqkv  = (const float*)d_in[6];
  const float* Wo    = (const float*)d_in[7];
  const float* bo    = (const float*)d_in[8];
  const float* ln2s  = (const float*)d_in[9];
  const float* ln2b  = (const float*)d_in[10];
  const float* Wfc   = (const float*)d_in[11];
  const float* bfc   = (const float*)d_in[12];
  const float* Wpr   = (const float*)d_in[13];
  const float* bpr   = (const float*)d_in[14];
  const float* lnfs  = (const float*)d_in[15];
  const float* lnfb  = (const float*)d_in[16];
  const float* Whead = (const float*)d_in[17];
  float* out = (float*)d_out;

  const long BT = (long)Bc * Tt;                 // 2048

  // ---- workspace carve-up
  float* x = (float*)d_ws;                                   // [BT,C] fp32 residual
  _Float16* h    = (_Float16*)(x + BT * Cc);                 // [BT,C]
  _Float16* qkv  = h   + BT * Cc;                            // [BT,3C]
  _Float16* y    = qkv + BT * 3 * Cc;                        // [BT,C]
  _Float16* u    = y   + BT * Cc;                            // [BT,FF]
  _Float16* vTb  = u   + BT * FFf;                           // [B*H*D, T]
  _Float16* wqT  = vTb + (long)Bc * Hh * Dd * Tt;            // [L,3C,C]
  _Float16* woT  = wqT + (long)Ll * 3 * Cc * Cc;             // [L,C,C]
  _Float16* wfT  = woT + (long)Ll * Cc * Cc;                 // [L,FF,C]
  _Float16* wpT  = wfT + (long)Ll * Cc * FFf;                // [L,C,FF]
  _Float16* whT  = wpT + (long)Ll * FFf * Cc;                // [V,C]

  // ---- weight conversion to transposed f16
  k_cvtT<<<dim3(Cc / 256, 3 * Cc, Ll), 256, 0, stream>>>(Wqkv, wqT, Cc, 3 * Cc);
  k_cvtT<<<dim3(Cc / 256, Cc, Ll),     256, 0, stream>>>(Wo,   woT, Cc, Cc);
  k_cvtT<<<dim3(Cc / 256, FFf, Ll),    256, 0, stream>>>(Wfc,  wfT, Cc, FFf);
  k_cvtT<<<dim3(FFf / 256, Cc, Ll),    256, 0, stream>>>(Wpr,  wpT, FFf, Cc);
  k_cvtT<<<dim3(Cc / 256, Vv, 1),      256, 0, stream>>>(Whead, whT, Cc, Vv);

  k_embed<<<(int)((BT * Cc + 255) / 256), 256, 0, stream>>>(idx, tok, pos, x);

  const dim3 blk(256);
  const int gy = (int)(BT / 256);                // 8
  for (int l = 0; l < Ll; ++l) {
    k_ln<<<(int)(BT / 8), 256, 0, stream>>>(x, ln1s + l * Cc, ln1b + l * Cc, h);
    k_gemm<0, true, _Float16><<<dim3(3 * Cc / 128, gy), blk, 0, stream>>>(
        wqT + (long)l * 3 * Cc * Cc, h, bqkv + (long)l * 3 * Cc, nullptr, qkv,
        (int)BT, 3 * Cc, Cc);
    k_vT<<<dim3(Tt / 256, Dd, Bc * Hh), 256, 0, stream>>>(qkv, vTb);
    k_attn<<<Bc * Hh * 64 / 8, 256, 0, stream>>>(qkv, vTb, y);
    k_gemm<1, true, float><<<dim3(Cc / 128, gy), blk, 0, stream>>>(
        woT + (long)l * Cc * Cc, y, bo + (long)l * Cc, x, x, (int)BT, Cc, Cc);
    k_ln<<<(int)(BT / 8), 256, 0, stream>>>(x, ln2s + l * Cc, ln2b + l * Cc, h);
    k_gemm<2, true, _Float16><<<dim3(FFf / 128, gy), blk, 0, stream>>>(
        wfT + (long)l * Cc * FFf, h, bfc + (long)l * FFf, nullptr, u,
        (int)BT, FFf, Cc);
    k_gemm<1, true, float><<<dim3(Cc / 128, gy), blk, 0, stream>>>(
        wpT + (long)l * FFf * Cc, u, bpr + (long)l * Cc, x, x, (int)BT, Cc, FFf);
  }

  k_ln<<<(int)(BT / 8), 256, 0, stream>>>(x, lnfs, lnfb, h);
  k_gemm<0, false, float><<<dim3((Vv + 127) / 128, gy), blk, 0, stream>>>(
      whT, h, nullptr, nullptr, out, (int)BT, Vv, Cc);
}
